// GraphAttn_68496138436782
// MI455X (gfx1250) — compile-verified, matrix operand order checked by reading, not run on hardware
//
#include <hip/hip_runtime.h>
#include <math.h>

typedef __attribute__((ext_vector_type(16))) __bf16 v16bf;
typedef __attribute__((ext_vector_type(8)))  float  v8f;

#define D  128
#define NH 8
#define HD 16

// Packed weight: per matrix, 8 jt * 4 c * 32 lanes * 16 halves = 16384 bf16 (32 KB)
#define WFRAG_HALVES 16384

// ---------------------------------------------------------------------------
// Kernel 0: pack a 128x128 f32 weight matrix into split-bf16 (hi, lo) pairs
// laid out exactly as the WMMA B-fragments consumed by the GEMMs:
//   dst[((jt*4 + c)*32 + lane)*16 + i]
// so the GEMMs do two contiguous b128 loads per fragment and ZERO cvt VALU.
// One 1024-thread block per matrix; thread = (jt,c,lane).
// ---------------------------------------------------------------------------
__global__ void __launch_bounds__(1024)
pack_weights_kernel(const float* __restrict__ W0, const float* __restrict__ W1,
                    const float* __restrict__ W2, const float* __restrict__ W3,
                    const float* __restrict__ W4, __bf16* __restrict__ dst)
{
    const float* W;
    switch (blockIdx.x) {
        case 0:  W = W0; break;
        case 1:  W = W1; break;
        case 2:  W = W2; break;
        case 3:  W = W3; break;
        default: W = W4; break;
    }
    __bf16* hi = dst + (size_t)blockIdx.x * 2 * WFRAG_HALVES;
    __bf16* lo = hi + WFRAG_HALVES;

    const int t    = threadIdx.x;          // 0..1023 == (jt*4+c)*32 + lane
    const int lane = t & 31;
    const int jc   = t >> 5;               // jt*4 + c
    const int jt   = jc >> 2;
    const int c    = jc & 3;
    const int row  = jt * 16 + (lane & 15);
    const int koff = (lane & 16) ? 8 : 0;  // ISA 7.12.2 K sub-block per lane half

    const float* p = W + (size_t)row * D + c * 32 + koff;
    v16bf h, l;
#pragma unroll
    for (int i = 0; i < 8; ++i) {
        float f = p[i];                    // halves 0..7  <- K = base..base+7
        __bf16 hb = (__bf16)f;
        h[i] = hb; l[i] = (__bf16)(f - (float)hb);
    }
#pragma unroll
    for (int i = 0; i < 8; ++i) {
        float f = p[16 + i];               // halves 8..15 <- K = base+16..base+23
        __bf16 hb = (__bf16)f;
        h[8 + i] = hb; l[8 + i] = (__bf16)(f - (float)hb);
    }
    *(v16bf*)(hi + (size_t)t * 16) = h;
    *(v16bf*)(lo + (size_t)t * 16) = l;
}

// ---------------------------------------------------------------------------
// Kernel 1: per-row LayerNorm + exact GELU.  One wave32 per row (128 = 4/lane).
// ---------------------------------------------------------------------------
__global__ void __launch_bounds__(256)
ln_gelu_kernel(const float* __restrict__ x,
               const float* __restrict__ scale,
               const float* __restrict__ bias,
               float* __restrict__ out, int nrows)
{
    const int wave = threadIdx.x >> 5;
    const int lane = threadIdx.x & 31;
    const int row  = blockIdx.x * (blockDim.x >> 5) + wave;
    if (row >= nrows) return;                      // wave-uniform guard

    const float4 v = *(const float4*)(x + (size_t)row * D + lane * 4);
    float s  = v.x + v.y + v.z + v.w;
    float ss = v.x*v.x + v.y*v.y + v.z*v.z + v.w*v.w;
#pragma unroll
    for (int m = 1; m < 32; m <<= 1) {
        s  += __shfl_xor(s,  m, 32);
        ss += __shfl_xor(ss, m, 32);
    }
    const float mean = s * (1.0f / D);
    const float var  = ss * (1.0f / D) - mean * mean;
    const float rstd = rsqrtf(var + 1e-5f);

    const float4 sc = *(const float4*)(scale + lane * 4);
    const float4 bi = *(const float4*)(bias  + lane * 4);
    float4 g;
    float y;
    y = (v.x - mean) * rstd * sc.x + bi.x; g.x = 0.5f * y * (1.0f + erff(y * 0.70710678118f));
    y = (v.y - mean) * rstd * sc.y + bi.y; g.y = 0.5f * y * (1.0f + erff(y * 0.70710678118f));
    y = (v.z - mean) * rstd * sc.z + bi.z; g.z = 0.5f * y * (1.0f + erff(y * 0.70710678118f));
    y = (v.w - mean) * rstd * sc.w + bi.w; g.w = 0.5f * y * (1.0f + erff(y * 0.70710678118f));
    *(float4*)(out + (size_t)row * D + lane * 4) = g;
}

// ---------------------------------------------------------------------------
// Shared GEMM tile machinery: out[16 rows, 128 cols] = (A16x128 @ W^T + bias)*s
// (+ add).  Split-bf16: hi*hi + hi*lo + lo*hi with f32 accumulate.
// ---------------------------------------------------------------------------
__device__ __forceinline__ void load_split_fragA(const float* __restrict__ p,
                                                 v16bf& hi, v16bf& lo)
{
    float f[16];
    float4 t;
    t = *(const float4*)(p +  0); f[0]=t.x;  f[1]=t.y;  f[2]=t.z;  f[3]=t.w;
    t = *(const float4*)(p +  4); f[4]=t.x;  f[5]=t.y;  f[6]=t.z;  f[7]=t.w;
    t = *(const float4*)(p + 16); f[8]=t.x;  f[9]=t.y;  f[10]=t.z; f[11]=t.w;
    t = *(const float4*)(p + 20); f[12]=t.x; f[13]=t.y; f[14]=t.z; f[15]=t.w;
#pragma unroll
    for (int i = 0; i < 16; ++i) {
        __bf16 h = (__bf16)f[i];
        hi[i] = h;
        lo[i] = (__bf16)(f[i] - (float)h);
    }
}

__device__ __forceinline__ void load_fragsA(const float* __restrict__ A, int m,
                                            int koff, v16bf ahi[4], v16bf alo[4])
{
    const float* arow = A + (size_t)m * D;
#pragma unroll
    for (int c = 0; c < 4; ++c)
        load_split_fragA(arow + c * 32 + koff, ahi[c], alo[c]);
}

__device__ __forceinline__ void gemm_rowtile(const v16bf ahi[4], const v16bf alo[4],
                                             const __bf16* __restrict__ Bhi,
                                             const __bf16* __restrict__ Blo,
                                             const float*  __restrict__ bias,
                                             const float*  __restrict__ add,
                                             float scl, float* __restrict__ out,
                                             int rt, int lane, int laneN, int hiHalf)
{
#pragma unroll
    for (int jt = 0; jt < 8; ++jt) {           // 8 column tiles of 16
        v8f acc = {};
#pragma unroll
        for (int c = 0; c < 4; ++c) {
            const size_t foff = (size_t)((jt * 4 + c) * 32 + lane) * 16;
            v16bf bhi = *(const v16bf*)(Bhi + foff);   // two b128 loads, no cvt
            v16bf blo = *(const v16bf*)(Blo + foff);
            acc = __builtin_amdgcn_wmma_f32_16x16x32_bf16(false, ahi[c], false, bhi,
                                                          (short)0, acc, false, false);
            acc = __builtin_amdgcn_wmma_f32_16x16x32_bf16(false, ahi[c], false, blo,
                                                          (short)0, acc, false, false);
            acc = __builtin_amdgcn_wmma_f32_16x16x32_bf16(false, alo[c], false, bhi,
                                                          (short)0, acc, false, false);
        }
        const int   col = jt * 16 + laneN;
        const float b   = bias ? bias[col] : 0.0f;
#pragma unroll
        for (int r = 0; r < 8; ++r) {
            const int    row = (rt << 4) + r + hiHalf;
            const size_t idx = (size_t)row * D + col;
            float v = (acc[r] + b) * scl;
            if (add) v += add[idx];
            out[idx] = v;
        }
    }
}

// Generic single GEMM (used for Ws with residual-add and Wo with skip-add).
__global__ void __launch_bounds__(256)
gemm128_kernel(const float*  __restrict__ A,
               const __bf16* __restrict__ Bhi, const __bf16* __restrict__ Blo,
               const float*  __restrict__ bias, const float* __restrict__ add,
               float scl, float* __restrict__ out, int nrows)
{
    const int wave  = threadIdx.x >> 5;
    const int lane  = threadIdx.x & 31;
    const int rt    = blockIdx.x * (blockDim.x >> 5) + wave;
    if (rt >= (nrows >> 4)) return;            // wave-uniform: EXEC all-1 inside

    const int laneN  = lane & 15;
    const int koff   = (lane & 16) ? 8 : 0;
    const int hiHalf = (lane & 16) ? 8 : 0;

    v16bf ahi[4], alo[4];
    load_fragsA(A, (rt << 4) + laneN, koff, ahi, alo);
    gemm_rowtile(ahi, alo, Bhi, Blo, bias, add, scl, out, rt, lane, laneN, hiHalf);
}

// Fused Q/K/V: dstruc fragments loaded+split ONCE for both Wq and Wk,
// then delta fragments for Wv.  288 WMMAs per wave, one launch instead of 3.
__global__ void __launch_bounds__(256)
qkv_kernel(const float*  __restrict__ dstruc, const float* __restrict__ delta,
           const __bf16* __restrict__ WqHi,   const __bf16* __restrict__ WqLo,
           const __bf16* __restrict__ WkHi,   const __bf16* __restrict__ WkLo,
           const __bf16* __restrict__ WvHi,   const __bf16* __restrict__ WvLo,
           const float*  __restrict__ bq,     const float* __restrict__ bk,
           const float*  __restrict__ bv,
           float* __restrict__ qry, float* __restrict__ keyf,
           float* __restrict__ valb, int nrows)
{
    const int wave  = threadIdx.x >> 5;
    const int lane  = threadIdx.x & 31;
    const int rt    = blockIdx.x * (blockDim.x >> 5) + wave;
    if (rt >= (nrows >> 4)) return;            // wave-uniform

    const int laneN  = lane & 15;
    const int koff   = (lane & 16) ? 8 : 0;
    const int hiHalf = (lane & 16) ? 8 : 0;
    const int m      = (rt << 4) + laneN;

    v16bf ahi[4], alo[4];
    load_fragsA(dstruc, m, koff, ahi, alo);    // shared by q and k
    gemm_rowtile(ahi, alo, WqHi, WqLo, bq, nullptr, 0.25f, qry,  rt, lane, laneN, hiHalf);
    gemm_rowtile(ahi, alo, WkHi, WkLo, bk, nullptr, 1.0f,  keyf, rt, lane, laneN, hiHalf);

    load_fragsA(delta, m, koff, ahi, alo);     // v path
    gemm_rowtile(ahi, alo, WvHi, WvLo, bv, nullptr, 1.0f,  valb, rt, lane, laneN, hiHalf);
}

// ---------------------------------------------------------------------------
// Kernel 3: edge attention.  One wave32 per edge; lane group of 4 = one head.
// attn[e,h] = ew[e] * sigmoid(<q[src,h,:], k[dst,h,:]>); agg[src] += attn*val[dst]
// ---------------------------------------------------------------------------
__global__ void __launch_bounds__(256)
edge_kernel(const float* __restrict__ qry, const float* __restrict__ keyf,
            const float* __restrict__ val, const float* __restrict__ ew,
            const int* __restrict__ eids, float* __restrict__ agg, int E)
{
    const int wave = threadIdx.x >> 5;
    const int lane = threadIdx.x & 31;
    const int e    = blockIdx.x * (blockDim.x >> 5) + wave;
    if (e >= E) return;

    const int   src = eids[e];
    const int   dst = eids[E + e];
    const float w   = ew[e];

    const float4 q = *(const float4*)(qry  + (size_t)src * D + lane * 4);
    const float4 k = *(const float4*)(keyf + (size_t)dst * D + lane * 4);
    float s = q.x * k.x + q.y * k.y + q.z * k.z + q.w * k.w;
    s += __shfl_xor(s, 1, 32);                 // reduce over the 4 lanes of a head
    s += __shfl_xor(s, 2, 32);
    const float attn = w / (1.0f + __expf(-s));

    const float4 v = *(const float4*)(val + (size_t)dst * D + lane * 4);
    float* o = agg + (size_t)src * D + lane * 4;
    unsafeAtomicAdd(o + 0, attn * v.x);        // -> global_atomic_add_f32
    unsafeAtomicAdd(o + 1, attn * v.y);
    unsafeAtomicAdd(o + 2, attn * v.z);
    unsafeAtomicAdd(o + 3, attn * v.w);
}

// ---------------------------------------------------------------------------
extern "C" void kernel_launch(void* const* d_in, const int* in_sizes, int n_in,
                              void* d_out, int out_size, void* d_ws, size_t ws_size,
                              hipStream_t stream)
{
    const float* in_feats = (const float*)d_in[0];
    const float* struc    = (const float*)d_in[1];
    const float* eweights = (const float*)d_in[2];
    const float* ln_scale = (const float*)d_in[3];
    const float* ln_bias  = (const float*)d_in[4];
    const float* Ws       = (const float*)d_in[5];
    const float* bs       = (const float*)d_in[6];
    const float* Wq       = (const float*)d_in[7];
    const float* bq       = (const float*)d_in[8];
    const float* Wk       = (const float*)d_in[9];
    const float* bk       = (const float*)d_in[10];
    const float* Wv       = (const float*)d_in[11];
    const float* bv       = (const float*)d_in[12];
    const float* Wo       = (const float*)d_in[13];
    const int*   eids     = (const int*)d_in[14];

    const int    N  = in_sizes[0] / D;
    const int    E  = in_sizes[2];
    const size_t nd = (size_t)N * D;

    float* delta  = (float*)d_ws;       // gelu(LN(in_feats))
    float* dstruc = delta  + nd;        // delta + struc @ Ws^T + bs
    float* qry    = dstruc + nd;
    float* keyf   = qry    + nd;
    float* valb   = keyf   + nd;
    float* agg    = valb   + nd;
    __bf16* wpack = (__bf16*)(agg + nd);   // 5 matrices * 2 * 16384 bf16 = 320 KB

    hipMemsetAsync(agg, 0, nd * sizeof(float), stream);

    // Pack all five weight matrices into split-bf16 WMMA B-fragments (once).
    pack_weights_kernel<<<5, 1024, 0, stream>>>(Ws, Wq, Wk, Wv, Wo, wpack);

    const __bf16* WsHi = wpack + 0 * 2 * WFRAG_HALVES;
    const __bf16* WqHi = wpack + 1 * 2 * WFRAG_HALVES;
    const __bf16* WkHi = wpack + 2 * 2 * WFRAG_HALVES;
    const __bf16* WvHi = wpack + 3 * 2 * WFRAG_HALVES;
    const __bf16* WoHi = wpack + 4 * 2 * WFRAG_HALVES;

    const int wpb = 8;                  // waves per 256-thread block
    dim3 blk(256);

    ln_gelu_kernel<<<(N + wpb - 1) / wpb, blk, 0, stream>>>(
        in_feats, ln_scale, ln_bias, delta, N);

    const int numRT = N / 16;
    const int gblk  = (numRT + wpb - 1) / wpb;

    // dstruc = delta + struc @ Ws^T + bs
    gemm128_kernel<<<gblk, blk, 0, stream>>>(struc, WsHi, WsHi + WFRAG_HALVES,
                                             bs, delta, 1.0f, dstruc, N);
    // q / k / v fused
    qkv_kernel<<<gblk, blk, 0, stream>>>(dstruc, delta,
                                         WqHi, WqHi + WFRAG_HALVES,
                                         WkHi, WkHi + WFRAG_HALVES,
                                         WvHi, WvHi + WFRAG_HALVES,
                                         bq, bk, bv, qry, keyf, valb, N);

    edge_kernel<<<(E + wpb - 1) / wpb, blk, 0, stream>>>(
        qry, keyf, valb, eweights, eids, agg, E);

    // out = agg @ Wo^T + in_feats (skip connection)
    gemm128_kernel<<<gblk, blk, 0, stream>>>(agg, WoHi, WoHi + WFRAG_HALVES,
                                             nullptr, in_feats, 1.0f,
                                             (float*)d_out, N);
}
